// CausalSelfAttention_66855460930270
// MI455X (gfx1250) — matrix operand order repaired and checked by reading, not compiled, not run on hardware
//
#include <hip/hip_runtime.h>

// ---------------- types ----------------
typedef __attribute__((ext_vector_type(4)))  unsigned u32x4;
typedef __attribute__((ext_vector_type(16))) __bf16   v16bf;
typedef __attribute__((ext_vector_type(8)))  float    v8f;

struct U32x8 { u32x4 lo, hi; };

static __device__ inline v16bf make_frag(u32x4 a, u32x4 b) {
    U32x8 t{a, b};
    return __builtin_bit_cast(v16bf, t);
}

static __device__ inline unsigned short f2bf(float f) {
    unsigned u = __builtin_bit_cast(unsigned, f);
    unsigned r = u + 0x7FFFu + ((u >> 16) & 1u);   // round-to-nearest-even
    return (unsigned short)(r >> 16);
}

// CDNA5 async global->LDS copy (ASYNCcnt-tracked), 16B per lane.
// LDS byte offset = low 32 bits of the generic pointer to __shared__.
static __device__ inline void async_b128(unsigned lds_off, const unsigned short* g) {
    asm volatile("global_load_async_to_lds_b128 %0, %1, off"
                 :: "v"(lds_off), "v"(g)
                 : "memory");
}
static __device__ inline void wait_async0() {
    asm volatile("s_wait_asynccnt 0x0" ::: "memory");
}

// ---------------- constants ----------------
// B=4, T=2048, C=1024, H=16, D=64, BT=8192

// ---------------- fp32 -> bf16 convert ----------------
__global__ void cvt_f32_bf16(const float* __restrict__ in,
                             unsigned short* __restrict__ out, int n) {
    int i = blockIdx.x * blockDim.x + threadIdx.x;
    if (i < n) out[i] = f2bf(in[i]);
}

// ---------------- GEMM: [Mx1024] bf16 @ [1024x1024] bf16 + bias ----------------
// Double-buffered LDS fed by async global->LDS copies; K-step 64.
// mode 0: out fp32 plain [row*1024+col]
// mode 1: out bf16 scattered to [B,H,T,D]   (Q, V)
// mode 2: out bf16 scattered to [B,H,D,T]   (K transposed)
__global__ __launch_bounds__(256) void gemm_bf16(
    const unsigned short* __restrict__ A,   // [M x 1024] bf16, row-major
    const unsigned short* __restrict__ W,   // [1024 x 1024] bf16, row-major [in,out]
    const float* __restrict__ bias,         // [1024] fp32
    void* __restrict__ outp, int mode) {
    __shared__ alignas(16) unsigned short As[2][128 * 72];   // 128 rows x K=64, pad->72
    __shared__ alignas(16) unsigned short Bs[2][64 * 136];   // 64 rows (K) x N=128, pad->136

    const int tid  = threadIdx.x;
    const int lane = tid & 31;
    const int w    = tid >> 5;          // 0..7
    const int wm   = w & 3;             // wave M index (0..3), 32 rows each
    const int wn   = w >> 2;            // wave N index (0..1), 64 cols each
    const int ln16 = lane & 15;
    const int hi8  = (lane < 16) ? 0 : 8;
    const int m0   = blockIdx.y * 128;
    const int n0   = blockIdx.x * 128;

    v8f acc[2][4];
#pragma unroll
    for (int a = 0; a < 2; a++)
#pragma unroll
        for (int b = 0; b < 4; b++) acc[a][b] = {};

    // async-copy work split: each thread moves 2x32B of A and 2x32B of B per K-step
    const int a_row = tid >> 2, a_seg = tid & 3;   // A tile: 128 rows x 4 segs of 16 elems
    const int b_row = tid >> 3, b_seg = tid & 7;   // B tile: 64 rows  x 8 segs of 16 elems

    auto issue_tile = [&](int p, int k0) {
        const unsigned sA = (unsigned)(size_t)(&As[p][0]);
        const unsigned sB = (unsigned)(size_t)(&Bs[p][0]);
#pragma unroll
        for (int u = 0; u < 2; u++) {
            const int r = a_row + u * 64;
            const unsigned short* g = A + (size_t)(m0 + r) * 1024 + k0 + a_seg * 16;
            const unsigned d = sA + (unsigned)(r * 72 + a_seg * 16) * 2u;
            async_b128(d, g);
            async_b128(d + 16u, g + 8);
        }
#pragma unroll
        for (int u = 0; u < 2; u++) {
            const int r = b_row + u * 32;
            const unsigned short* g = W + (size_t)(k0 + r) * 1024 + n0 + b_seg * 16;
            const unsigned d = sB + (unsigned)(r * 136 + b_seg * 16) * 2u;
            async_b128(d, g);
            async_b128(d + 16u, g + 8);
        }
    };

    // prologue: fill buffer 0
    issue_tile(0, 0);
    wait_async0();
    __syncthreads();

    int p = 0;
    for (int k0 = 0; k0 < 1024; k0 += 64) {
        if (k0 + 64 < 1024) issue_tile(p ^ 1, k0 + 64);   // prefetch next tile (async)
#pragma unroll
        for (int kk = 0; kk < 2; kk++) {
#pragma unroll
            for (int mf = 0; mf < 2; mf++) {
                const unsigned short* ap =
                    &As[p][(wm * 32 + mf * 16 + ln16) * 72 + kk * 32 + hi8];
                v16bf afrag = make_frag(*(const u32x4*)ap, *(const u32x4*)(ap + 16));
#pragma unroll
                for (int nf = 0; nf < 4; nf++) {
                    const unsigned short* bp =
                        &Bs[p][(kk * 32 + hi8) * 136 + wn * 64 + nf * 16 + ln16];
                    v16bf bfrag;
#pragma unroll
                    for (int s = 0; s < 8; s++) {
                        bfrag[s]     = __builtin_bit_cast(__bf16, bp[s * 136]);
                        bfrag[8 + s] = __builtin_bit_cast(__bf16, bp[(16 + s) * 136]);
                    }
                    acc[mf][nf] = __builtin_amdgcn_wmma_f32_16x16x32_bf16(
                        false, afrag, false, bfrag, (short)0, acc[mf][nf], false, false);
                }
            }
        }
        wait_async0();       // our writes into buf p^1 have landed
        __syncthreads();     // everyone done reading buf p, everyone's copies visible
        p ^= 1;
    }

#pragma unroll
    for (int mf = 0; mf < 2; mf++) {
#pragma unroll
        for (int nf = 0; nf < 4; nf++) {
            const int col = n0 + wn * 64 + nf * 16 + ln16;
            const float bv = bias[col];
#pragma unroll
            for (int i = 0; i < 8; i++) {
                const int row = m0 + wm * 32 + mf * 16 + ((lane < 16) ? i : 8 + i);
                const float val = acc[mf][nf][i] + bv;
                if (mode == 0) {
                    ((float*)outp)[(size_t)row * 1024 + col] = val;
                } else {
                    const unsigned short hv = f2bf(val);
                    const int b = row >> 11, t = row & 2047;   // T = 2048
                    const int h = col >> 6,  dd = col & 63;    // D = 64
                    unsigned short* o = (unsigned short*)outp;
                    if (mode == 1)
                        o[((size_t)((b << 4) + h) * 2048 + t) * 64 + dd] = hv;   // [B,H,T,D]
                    else
                        o[((size_t)((b << 4) + h) * 64 + dd) * 2048 + t] = hv;   // [B,H,D,T]
                }
            }
        }
    }
}

// ---------------- flash attention ----------------
// Q: [BH, T, 64] bf16   Kt: [BH, 64, T] bf16   V: [BH, T, 64] bf16
// Y: [B, T, 1024] bf16
__global__ __launch_bounds__(128) void attn_fwd(
    const unsigned short* __restrict__ Q,
    const unsigned short* __restrict__ Kt,
    const unsigned short* __restrict__ V,
    unsigned short* __restrict__ Y) {
    __shared__ alignas(16) unsigned short P[4][16 * 72];   // per-wave 16x64, pad->72

    const int tid  = threadIdx.x;
    const int lane = tid & 31;
    const int w    = tid >> 5;            // 0..3, each wave = 16 query rows
    const int ln16 = lane & 15;
    const int hi8  = (lane < 16) ? 0 : 8;
    const int bh = blockIdx.y;
    const int b  = bh >> 4, h = bh & 15;
    const int q0 = blockIdx.x * 64;
    const int qr = q0 + w * 16;           // wave's first query row

    // Q fragments (16 rows x 64 dims = 2 K-chunks), resident all loop
    v16bf aQ[2];
    {
        const unsigned short* qbase = Q + ((size_t)bh * 2048 + qr + ln16) * 64;
#pragma unroll
        for (int kk = 0; kk < 2; kk++) {
            const unsigned short* p = qbase + kk * 32 + hi8;
            aQ[kk] = make_frag(*(const u32x4*)p, *(const u32x4*)(p + 16));
        }
    }

    float m[8], l[8];
#pragma unroll
    for (int i = 0; i < 8; i++) { m[i] = -3.0e38f; l[i] = 0.0f; }
    v8f accO[4];
#pragma unroll
    for (int f = 0; f < 4; f++) accO[f] = {};

    unsigned short* Pl = P[w];

    for (int key0 = 0; key0 <= q0; key0 += 64) {
        // ---- S = Q @ K^T  (16 x 64 keys) ----
        v8f s[4];
#pragma unroll
        for (int nf = 0; nf < 4; nf++) s[nf] = {};
#pragma unroll
        for (int kk = 0; kk < 2; kk++) {
            const unsigned short* kb = Kt + ((size_t)bh * 64 + kk * 32 + hi8) * 2048 + key0;
#pragma unroll
            for (int nf = 0; nf < 4; nf++) {
                const unsigned short* p = kb + nf * 16 + ln16;
                v16bf bfrag;
#pragma unroll
                for (int s = 0; s < 8; s++) {
                    bfrag[s]     = __builtin_bit_cast(__bf16, p[s * 2048]);
                    bfrag[8 + s] = __builtin_bit_cast(__bf16, p[(16 + s) * 2048]);
                }
                s[nf] = __builtin_amdgcn_wmma_f32_16x16x32_bf16(
                    false, aQ[kk], false, bfrag, (short)0, s[nf], false, false);
            }
        }
        // ---- scale + causal mask + online softmax ----
#pragma unroll
        for (int i = 0; i < 8; i++) {
            const int r  = (lane < 16) ? i : 8 + i;
            const int qa = qr + r;
            float mx = -3.0e38f;
#pragma unroll
            for (int nf = 0; nf < 4; nf++) {
                float sv = s[nf][i] * 0.125f;               // 1/sqrt(64)
                const int key = key0 + nf * 16 + ln16;
                sv = (key <= qa) ? sv : -3.0e38f;
                s[nf][i] = sv;
                mx = fmaxf(mx, sv);
            }
            mx = fmaxf(mx, __shfl_xor(mx, 1));
            mx = fmaxf(mx, __shfl_xor(mx, 2));
            mx = fmaxf(mx, __shfl_xor(mx, 4));
            mx = fmaxf(mx, __shfl_xor(mx, 8));
            const float mn = fmaxf(m[i], mx);
            const float sc = __expf(m[i] - mn);
            float rs = 0.0f;
#pragma unroll
            for (int nf = 0; nf < 4; nf++) {
                const float pv = __expf(s[nf][i] - mn);
                s[nf][i] = pv;
                rs += pv;
            }
            rs += __shfl_xor(rs, 1);
            rs += __shfl_xor(rs, 2);
            rs += __shfl_xor(rs, 4);
            rs += __shfl_xor(rs, 8);
            l[i] = l[i] * sc + rs;
            m[i] = mn;
#pragma unroll
            for (int f = 0; f < 4; f++) accO[f][i] *= sc;
            // C-layout -> A-layout reshuffle via per-wave LDS (same-wave LDS is in-order)
#pragma unroll
            for (int nf = 0; nf < 4; nf++) Pl[r * 72 + nf * 16 + ln16] = f2bf(s[nf][i]);
        }
        // ---- O += P @ V ----
#pragma unroll
        for (int kk = 0; kk < 2; kk++) {
            const unsigned short* pp = Pl + ln16 * 72 + kk * 32 + hi8;
            v16bf aP = make_frag(*(const u32x4*)pp, *(const u32x4*)(pp + 16));
            const unsigned short* vb = V + ((size_t)bh * 2048 + key0 + kk * 32 + hi8) * 64;
#pragma unroll
            for (int f = 0; f < 4; f++) {
                const unsigned short* p = vb + f * 16 + ln16;
                v16bf bfrag;
#pragma unroll
                for (int s = 0; s < 8; s++) {
                    bfrag[s]     = __builtin_bit_cast(__bf16, p[s * 64]);
                    bfrag[8 + s] = __builtin_bit_cast(__bf16, p[(16 + s) * 64]);
                }
                accO[f] = __builtin_amdgcn_wmma_f32_16x16x32_bf16(
                    false, aP, false, bfrag, (short)0, accO[f], false, false);
            }
        }
    }

    // ---- write y (bf16, [B,T,C] with c = h*64 + d) ----
#pragma unroll
    for (int f = 0; f < 4; f++) {
#pragma unroll
        for (int i = 0; i < 8; i++) {
            const int r = (lane < 16) ? i : 8 + i;
            const float val = accO[f][i] / l[i];
            const int row = qr + r;
            Y[((size_t)b * 2048 + row) * 1024 + h * 64 + f * 16 + ln16] = f2bf(val);
        }
    }
}

// ---------------- launch ----------------
extern "C" void kernel_launch(void* const* d_in, const int* in_sizes, int n_in,
                              void* d_out, int out_size, void* d_ws, size_t ws_size,
                              hipStream_t stream) {
    const float* x  = (const float*)d_in[0];
    const float* Wq = (const float*)d_in[1];
    const float* bq = (const float*)d_in[2];
    const float* Wk = (const float*)d_in[3];
    const float* bk = (const float*)d_in[4];
    const float* Wv = (const float*)d_in[5];
    const float* bv = (const float*)d_in[6];
    const float* Wp = (const float*)d_in[7];
    const float* bp = (const float*)d_in[8];

    char* ws = (char*)d_ws;
    // byte offsets (all 256-aligned). yb aliases xb (x consumed before attn writes).
    unsigned short* xb  = (unsigned short*)(ws + 0);          // 16 MB: x bf16 [8192,1024]
    unsigned short* yb  = xb;                                  // reuse after projections
    unsigned short* wqb = (unsigned short*)(ws + 16777216);    // 2 MB each
    unsigned short* wkb = (unsigned short*)(ws + 18874368);
    unsigned short* wvb = (unsigned short*)(ws + 20971520);
    unsigned short* wpb = (unsigned short*)(ws + 23068672);
    unsigned short* qb  = (unsigned short*)(ws + 25165824);    // 16 MB [BH,T,64]
    unsigned short* ktb = (unsigned short*)(ws + 41943040);    // 16 MB [BH,64,T]
    unsigned short* vb_ = (unsigned short*)(ws + 58720256);    // 16 MB [BH,T,64]
    // total = 75,497,472 bytes

    const int NX = 8192 * 1024;   // x elements
    const int NW = 1024 * 1024;   // weight elements
    cvt_f32_bf16<<<(NX + 255) / 256, 256, 0, stream>>>(x,  xb,  NX);
    cvt_f32_bf16<<<(NW + 255) / 256, 256, 0, stream>>>(Wq, wqb, NW);
    cvt_f32_bf16<<<(NW + 255) / 256, 256, 0, stream>>>(Wk, wkb, NW);
    cvt_f32_bf16<<<(NW + 255) / 256, 256, 0, stream>>>(Wv, wvb, NW);
    cvt_f32_bf16<<<(NW + 255) / 256, 256, 0, stream>>>(Wp, wpb, NW);

    dim3 gg(1024 / 128, 8192 / 128);   // (8, 64)
    gemm_bf16<<<gg, 256, 0, stream>>>(xb, wqb, bq, (void*)qb,  1);
    gemm_bf16<<<gg, 256, 0, stream>>>(xb, wkb, bk, (void*)ktb, 2);
    gemm_bf16<<<gg, 256, 0, stream>>>(xb, wvb, bv, (void*)vb_, 1);

    attn_fwd<<<dim3(2048 / 64, 4 * 16), 128, 0, stream>>>(qb, ktb, vb_, yb);

    gemm_bf16<<<gg, 256, 0, stream>>>(yb, wpb, bp, d_out, 0);
}
